// SA_13735305413331
// MI455X (gfx1250) — matrix operand order repaired and checked by reading, not compiled, not run on hardware
//
#include <hip/hip_runtime.h>
#include <hip/hip_bf16.h>
#include <math.h>

// ---------------------------------------------------------------------------
// Types for CDNA5 WMMA (gfx1250, wave32)
// ---------------------------------------------------------------------------
typedef __bf16 bf16;
typedef __attribute__((ext_vector_type(16))) __bf16 v16bf;
typedef __attribute__((ext_vector_type(8)))  __bf16 v8bf;
typedef __attribute__((ext_vector_type(8)))  float  v8f;

#define ACT_NONE 0
#define ACT_RELU 1
#define ACT_ELU  2

// ---------------------------------------------------------------------------
// CDNA5 async global->LDS copy (ASYNCcnt path, no VGPR roundtrip)
// ---------------------------------------------------------------------------
__device__ __forceinline__ void async_copy_b128(const bf16* gsrc, bf16* lds_dst) {
    unsigned lds = (unsigned)(unsigned long long)lds_dst;  // flat low 32 bits = LDS offset
    asm volatile("global_load_async_to_lds_b128 %0, %1, off"
                 :: "v"(lds), "v"(gsrc) : "memory");
}
__device__ __forceinline__ void wait_async0() {
    asm volatile("s_wait_asynccnt 0x0" ::: "memory");
}

// ---------------------------------------------------------------------------
// cast fp32 -> bf16 with row/col zero padding (pads K to mult of 32)
// ---------------------------------------------------------------------------
__global__ void cast_pad_bf16(const float* __restrict__ src, bf16* __restrict__ dst,
                              int R, int C, int Rp, int Cp) {
    size_t i = (size_t)blockIdx.x * blockDim.x + threadIdx.x;
    size_t total = (size_t)Rp * Cp;
    if (i >= total) return;
    int r = (int)(i / Cp);
    int c = (int)(i % Cp);
    float v = (r < R && c < C) ? src[(size_t)r * C + c] : 0.0f;
    dst[i] = (bf16)v;
}

// ---------------------------------------------------------------------------
// WMMA bf16 GEMM:  C = act(A[M,K] @ B[K,N] + bias),  K mult of 32, N mult of 64
// Block: 256 thr (8 waves). Tile 128(M) x 64(N). Wave -> 32x32 (2x2 WMMA).
// Double-buffered LDS; A staged via global_load_async_to_lds_b128; B staged
// transposed ([n][k], stride 40 bf16 keeps 16B alignment for ds_load_b128).
// ---------------------------------------------------------------------------
__device__ __forceinline__ v16bf frag_a_from_lds(const bf16* base) {
    // A 16x32 layout: lane m=l&15; kbase=(l<16)?0:16; elems K[kbase..kbase+15] contiguous
    v8bf lo = *(const v8bf*)(base);
    v8bf hi = *(const v8bf*)(base + 8);
    v16bf r;
#pragma unroll
    for (int i = 0; i < 8; ++i) { r[i] = lo[i]; r[8 + i] = hi[i]; }
    return r;
}

__device__ __forceinline__ v16bf frag_b_from_lds(const bf16* base) {
    // B 32x16 layout: lane n=l&15; ks=(l<16)?0:16; elems K[ks..ks+15] contiguous in Bs[n][k]
    v8bf lo = *(const v8bf*)(base);
    v8bf hi = *(const v8bf*)(base + 8);
    v16bf r;
#pragma unroll
    for (int i = 0; i < 8; ++i) { r[i] = lo[i]; r[8 + i] = hi[i]; }
    return r;
}

template <int ACT>
__global__ __launch_bounds__(256)
void gemm_bf16_wmma(const bf16* __restrict__ A, int lda,
                    const bf16* __restrict__ B, const float* __restrict__ bias,
                    float* __restrict__ Cf, bf16* __restrict__ Cbf,
                    int M, int N, int K) {
    __shared__ bf16 As[2][128 * 40];
    __shared__ bf16 Bs[2][64 * 40];

    const int tid   = threadIdx.x;
    const int lane  = tid & 31;
    const int wave  = tid >> 5;
    const int waveM = wave >> 1;   // 0..3
    const int waveN = wave & 1;    // 0..1
    const int mBase = blockIdx.y * 128;
    const int nBase = blockIdx.x * 64;

    v8f acc[2][2];
#pragma unroll
    for (int a = 0; a < 2; ++a)
#pragma unroll
        for (int b = 0; b < 2; ++b)
#pragma unroll
            for (int v = 0; v < 8; ++v) acc[a][b][v] = 0.0f;

    // A-tile staging: 128x32 bf16, 2x b128 async copies per thread.
    // OOB rows are clamped (not zeroed): C row m depends only on A row m, and
    // OOB C rows are never stored, so duplicated data is harmless.
    auto stageA = [&](int buf, int k0) {
        int r = tid >> 1, h = tid & 1;
        int gm = mBase + r;
        if (gm >= M) gm = M - 1;
        const bf16* src = A + (size_t)gm * lda + k0 + h * 16;
        bf16* dst = &As[buf][r * 40 + h * 16];
        async_copy_b128(src, dst);
        async_copy_b128(src + 8, dst + 8);
    };
    // B-tile staging: 32x64 loaded row-major, scattered transposed into Bs[n][k].
    auto stageB = [&](int buf, int k0) {
        int kk = tid >> 3, ng = tid & 7;
        const bf16* src = B + (size_t)(k0 + kk) * N + nBase + ng * 8;
        v8bf v = *(const v8bf*)src;
#pragma unroll
        for (int i = 0; i < 8; ++i) Bs[buf][(ng * 8 + i) * 40 + kk] = v[i];
    };

    const int lm  = lane & 15;
    const int kbA = (lane < 16) ? 0 : 16;   // A contiguous pattern
    const int ksB = (lane < 16) ? 0 : 16;   // B contiguous pattern
    const int nk  = K >> 5;

    stageA(0, 0);
    stageB(0, 0);

    for (int i = 0; i < nk; ++i) {
        wait_async0();          // this wave's async tile-i writes have landed
        __syncthreads();        // all waves' tile-i staging visible
        const int cur = i & 1;
        if (i + 1 < nk) {       // prefetch next tile into the other buffer
            stageA(cur ^ 1, (i + 1) * 32);
            stageB(cur ^ 1, (i + 1) * 32);
        }

        const int mrow = waveM * 32;
        const int ncol = waveN * 32;
        v16bf a0 = frag_a_from_lds(&As[cur][(mrow + lm) * 40 + kbA]);
        v16bf a1 = frag_a_from_lds(&As[cur][(mrow + 16 + lm) * 40 + kbA]);
        v16bf b0 = frag_b_from_lds(&Bs[cur][(ncol + lm) * 40 + ksB]);
        v16bf b1 = frag_b_from_lds(&Bs[cur][(ncol + 16 + lm) * 40 + ksB]);

        acc[0][0] = __builtin_amdgcn_wmma_f32_16x16x32_bf16(false, a0, false, b0, (short)0, acc[0][0], false, false);
        acc[0][1] = __builtin_amdgcn_wmma_f32_16x16x32_bf16(false, a0, false, b1, (short)0, acc[0][1], false, false);
        acc[1][0] = __builtin_amdgcn_wmma_f32_16x16x32_bf16(false, a1, false, b0, (short)0, acc[1][0], false, false);
        acc[1][1] = __builtin_amdgcn_wmma_f32_16x16x32_bf16(false, a1, false, b1, (short)0, acc[1][1], false, false);

        __syncthreads();        // readers of buf(cur) done before it is restaged
    }

    // ---- epilogue: C/D layout (lane 0-15: M=v; lane 16-31: M=8+v; N=lane&15) ----
    const int mHalf = (lane >= 16) ? 8 : 0;
#pragma unroll
    for (int tm = 0; tm < 2; ++tm) {
#pragma unroll
        for (int tn = 0; tn < 2; ++tn) {
            int n = nBase + waveN * 32 + tn * 16 + (lane & 15);
            float bv = bias ? bias[n] : 0.0f;
#pragma unroll
            for (int v = 0; v < 8; ++v) {
                int gm = mBase + waveM * 32 + tm * 16 + mHalf + v;
                if (gm < M) {
                    float val = acc[tm][tn][v] + bv;
                    if (ACT == ACT_RELU) val = fmaxf(val, 0.0f);
                    if (ACT == ACT_ELU)  val = (val > 0.0f) ? val : (__expf(val) - 1.0f);
                    if (Cf)  Cf[(size_t)gm * N + n]  = val;
                    if (Cbf) Cbf[(size_t)gm * N + n] = (bf16)val;
                }
            }
        }
    }
}

// ---------------------------------------------------------------------------
// GAT helpers
// ---------------------------------------------------------------------------
__device__ __forceinline__ void atomicMaxF(float* addr, float val) {
    // classic int-bit trick: works across mixed signs
    if (val >= 0.0f) atomicMax((int*)addr, __float_as_int(val));
    else             atomicMin((unsigned int*)addr, __float_as_uint(val));
}

// per-node attention scores: as[i] = xW[i].a_src ; ad[i] = xW[i].a_dst (wave per node)
__global__ void attn_scores(const float* __restrict__ xw, const float* __restrict__ a_src,
                            const float* __restrict__ a_dst, float* __restrict__ as_,
                            float* __restrict__ ad_, int n, int F) {
    int node = blockIdx.x * 8 + (threadIdx.x >> 5);
    int lane = threadIdx.x & 31;
    if (node >= n) return;
    const float* row = xw + (size_t)node * F;
    float s = 0.0f, d = 0.0f;
    for (int f = lane; f < F; f += 32) { float x = row[f]; s += x * a_src[f]; d += x * a_dst[f]; }
#pragma unroll
    for (int off = 16; off; off >>= 1) { s += __shfl_xor(s, off, 32); d += __shfl_xor(d, off, 32); }
    if (lane == 0) { as_[node] = s; ad_[node] = d; }
}

__global__ void node_init(float* __restrict__ mx, float* __restrict__ den, int n) {
    int i = blockIdx.x * blockDim.x + threadIdx.x;
    if (i < n) { mx[i] = -3.0e38f; den[i] = 0.0f; }
}

// pass 1: e = leaky_relu(as[src]+ad[dst]); segment max over dst (self-loops appended)
__global__ void edge_logits(const int* __restrict__ ei, int E, int n,
                            const float* __restrict__ as_, const float* __restrict__ ad_,
                            float* __restrict__ ebuf, float* __restrict__ mx) {
    int i = blockIdx.x * blockDim.x + threadIdx.x;
    if (i >= E + n) return;
    int s, t;
    if (i < E) { s = ei[i]; t = ei[E + i]; } else { s = t = i - E; }
    float e = as_[s] + ad_[t];
    e = (e > 0.0f) ? e : 0.2f * e;
    ebuf[i] = e;
    atomicMaxF(&mx[t], e);
}

// pass 2: ex = exp(e - m[dst]); segment sum
__global__ void edge_expsum(const int* __restrict__ ei, int E, int n,
                            float* __restrict__ ebuf, const float* __restrict__ mx,
                            float* __restrict__ den) {
    int i = blockIdx.x * blockDim.x + threadIdx.x;
    if (i >= E + n) return;
    int t = (i < E) ? ei[E + i] : (i - E);
    float ex = __expf(ebuf[i] - mx[t]);
    ebuf[i] = ex;
    atomicAdd(&den[t], ex);
}

// pass 3: out[dst] += (ex/den[dst]) * xW[src]   (one wave32 per edge)
__global__ void edge_aggregate(const int* __restrict__ ei, int E, int n,
                               const float* __restrict__ ebuf, const float* __restrict__ den,
                               const float* __restrict__ xw, float* __restrict__ agg, int F) {
    int wid  = (blockIdx.x * blockDim.x + threadIdx.x) >> 5;
    int lane = threadIdx.x & 31;
    if (wid >= E + n) return;
    int s, t;
    if (wid < E) { s = ei[wid]; t = ei[E + wid]; } else { s = t = wid - E; }
    float alpha = ebuf[wid] / den[t];
    const float* xs = xw  + (size_t)s * F;
    float*       od = agg + (size_t)t * F;
    for (int f = lane; f < F; f += 32) atomicAdd(&od[f], alpha * xs[f]);
}

// GAT tail: relu(agg + b) -> bf16 (next GEMM A operand)
__global__ void bias_relu_cast(const float* __restrict__ agg, const float* __restrict__ b,
                               bf16* __restrict__ obf, int n, int F) {
    size_t i = (size_t)blockIdx.x * blockDim.x + threadIdx.x;
    if (i >= (size_t)n * F) return;
    int f = (int)(i % F);
    float v = agg[i] + b[f];
    v = fmaxf(v, 0.0f);
    obf[i] = (bf16)v;
}

// pair gather + concat to [B,512] bf16
__global__ void gather_concat(const bf16* __restrict__ dfeat, const bf16* __restrict__ cfeat,
                              const int* __restrict__ di, const int* __restrict__ ci,
                              bf16* __restrict__ comb, int B) {
    int i = blockIdx.x * blockDim.x + threadIdx.x;
    if (i >= B * 512) return;
    int b = i >> 9, f = i & 511;
    comb[i] = (f < 256) ? dfeat[(size_t)di[b] * 256 + f]
                        : cfeat[(size_t)ci[b] * 256 + (f - 256)];
}

// final 512 -> 1 GEMV (wave per row)
__global__ void head_out(const float* __restrict__ h, const float* __restrict__ w,
                         const float* __restrict__ b, float* __restrict__ out, int B, int K) {
    int row  = (blockIdx.x * blockDim.x + threadIdx.x) >> 5;
    int lane = threadIdx.x & 31;
    if (row >= B) return;
    const float* hr = h + (size_t)row * K;
    float s = 0.0f;
    for (int k = lane; k < K; k += 32) s += hr[k] * w[k];
#pragma unroll
    for (int off = 16; off; off >>= 1) s += __shfl_xor(s, off, 32);
    if (lane == 0) out[row] = s + b[0];
}

// ---------------------------------------------------------------------------
// host-side launcher
// ---------------------------------------------------------------------------
static inline int cdiv(long long a, long long b) { return (int)((a + b - 1) / b); }

extern "C" void kernel_launch(void* const* d_in, const int* in_sizes, int n_in,
                              void* d_out, int out_size, void* d_ws, size_t ws_size,
                              hipStream_t stream) {
    (void)n_in; (void)out_size; (void)ws_size;

    const float* drug_x = (const float*)d_in[0];
    const float* cell_x = (const float*)d_in[1];
    const int*   d_ei   = (const int*)d_in[2];   // int32 (x64 disabled by default)
    const int*   c_ei   = (const int*)d_in[3];
    const int*   d_idx  = (const int*)d_in[4];
    const int*   c_idx  = (const int*)d_in[5];
    const float* dW1 = (const float*)d_in[6];  const float* db1 = (const float*)d_in[7];
    const float* cW1 = (const float*)d_in[8];  const float* cb1 = (const float*)d_in[9];
    const float* cW2 = (const float*)d_in[10]; const float* cb2 = (const float*)d_in[11];
    const float* gdW = (const float*)d_in[12]; const float* gdas = (const float*)d_in[13];
    const float* gdad = (const float*)d_in[14]; const float* gdb = (const float*)d_in[15];
    const float* g1W = (const float*)d_in[16]; const float* g1as = (const float*)d_in[17];
    const float* g1ad = (const float*)d_in[18]; const float* g1b = (const float*)d_in[19];
    const float* g2W = (const float*)d_in[20]; const float* g2as = (const float*)d_in[21];
    const float* g2ad = (const float*)d_in[22]; const float* g2b = (const float*)d_in[23];
    const float* rW1 = (const float*)d_in[24]; const float* rb1 = (const float*)d_in[25];
    const float* rW2 = (const float*)d_in[26]; const float* rb2 = (const float*)d_in[27];
    const float* rW3 = (const float*)d_in[28]; const float* rb3 = (const float*)d_in[29];
    float* out = (float*)d_out;

    const int ND = 50000, FD = 128, NC = 10000, FC = 735, FCP = 736;
    const int ED = in_sizes[2] / 2, EC = in_sizes[3] / 2, NB = in_sizes[4];

    // ---- bump arena in d_ws ----
    char* p = (char*)d_ws;
    auto alloc = [&](size_t bytes) -> void* {
        void* r = (void*)p; p += (bytes + 255) & ~(size_t)255; return r;
    };
    bf16* dx_bf  = (bf16*)alloc((size_t)ND * FD  * 2);
    bf16* cx_bf  = (bf16*)alloc((size_t)NC * FCP * 2);
    bf16* dW1b = (bf16*)alloc((size_t)FD  * 256  * 2);
    bf16* cW1b = (bf16*)alloc((size_t)FCP * 1024 * 2);
    bf16* cW2b = (bf16*)alloc((size_t)1024 * 256 * 2);
    bf16* gdWb = (bf16*)alloc((size_t)256 * 256  * 2);
    bf16* g1Wb = (bf16*)alloc((size_t)256 * 1024 * 2);
    bf16* g2Wb = (bf16*)alloc((size_t)1024 * 256 * 2);
    bf16* rW1b = (bf16*)alloc((size_t)512 * 512  * 2);
    bf16* rW2b = (bf16*)alloc((size_t)512 * 512  * 2);
    bf16*  d_bf   = (bf16*) alloc((size_t)ND * 256 * 2);   // relu(drug_x@dW1+db1)
    bf16*  c1_bf  = (bf16*) alloc((size_t)NC * 1024 * 2);  // relu(cell_x@cW1+cb1)
    bf16*  c_bf   = (bf16*) alloc((size_t)NC * 256 * 2);   // relu(c1@cW2+cb2)
    float* dxw_f  = (float*)alloc((size_t)ND * 256 * 4);   // drug GAT x@W
    float* agg_d  = (float*)alloc((size_t)ND * 256 * 4);
    bf16*  dfin_bf= (bf16*) alloc((size_t)ND * 256 * 2);
    float* xw1_f  = (float*)alloc((size_t)NC * 1024 * 4);  // cell GAT1 x@W
    float* agg1   = (float*)alloc((size_t)NC * 1024 * 4);
    bf16*  c1g_bf = (bf16*) alloc((size_t)NC * 1024 * 2);
    float* xw2_f  = (float*)alloc((size_t)NC * 256 * 4);   // cell GAT2 x@W
    float* agg2   = (float*)alloc((size_t)NC * 256 * 4);
    bf16*  cfin_bf= (bf16*) alloc((size_t)NC * 256 * 2);
    float* as_d = (float*)alloc((size_t)ND * 4);
    float* ad_d = (float*)alloc((size_t)ND * 4);
    float* mx_d = (float*)alloc((size_t)ND * 4);
    float* dn_d = (float*)alloc((size_t)ND * 4);
    float* eb_d = (float*)alloc((size_t)(ED + ND) * 4);
    float* as_c = (float*)alloc((size_t)NC * 4);
    float* ad_c = (float*)alloc((size_t)NC * 4);
    float* mx_c = (float*)alloc((size_t)NC * 4);
    float* dn_c = (float*)alloc((size_t)NC * 4);
    float* eb_c = (float*)alloc((size_t)(EC + NC) * 4);
    bf16*  comb  = (bf16*) alloc((size_t)NB * 512 * 2);
    bf16*  h1_bf = (bf16*) alloc((size_t)NB * 512 * 2);
    float* h2_f  = (float*)alloc((size_t)NB * 512 * 4);

    const dim3 T(256);
    auto gemm = [&](int act, const bf16* A, int lda, const bf16* Bw, const float* bias,
                    float* Cf, bf16* Cbf, int M, int N, int K) {
        dim3 g(N / 64, cdiv(M, 128));
        if (act == ACT_NONE)
            gemm_bf16_wmma<ACT_NONE><<<g, T, 0, stream>>>(A, lda, Bw, bias, Cf, Cbf, M, N, K);
        else if (act == ACT_RELU)
            gemm_bf16_wmma<ACT_RELU><<<g, T, 0, stream>>>(A, lda, Bw, bias, Cf, Cbf, M, N, K);
        else
            gemm_bf16_wmma<ACT_ELU><<<g, T, 0, stream>>>(A, lda, Bw, bias, Cf, Cbf, M, N, K);
    };
    auto gat = [&](const int* ei, int E, int n, int F,
                   const float* asrc, const float* adst, const float* gbias,
                   float* xw, float* mx, float* dn, float* as_, float* ad_,
                   float* eb, float* agg, bf16* obf) {
        attn_scores<<<cdiv(n, 8), T, 0, stream>>>(xw, asrc, adst, as_, ad_, n, F);
        node_init<<<cdiv(n, 256), T, 0, stream>>>(mx, dn, n);
        hipMemsetAsync(agg, 0, (size_t)n * F * 4, stream);
        int Et = E + n;
        edge_logits   <<<cdiv(Et, 256), T, 0, stream>>>(ei, E, n, as_, ad_, eb, mx);
        edge_expsum   <<<cdiv(Et, 256), T, 0, stream>>>(ei, E, n, eb, mx, dn);
        edge_aggregate<<<cdiv(Et, 8),   T, 0, stream>>>(ei, E, n, eb, dn, xw, agg, F);
        bias_relu_cast<<<cdiv((long long)n * F, 256), T, 0, stream>>>(agg, gbias, obf, n, F);
    };

    // ---- bf16 casts (pad K to mult of 32 where needed: 735 -> 736) ----
    cast_pad_bf16<<<cdiv((long long)ND * FD, 256), T, 0, stream>>>(drug_x, dx_bf, ND, FD, ND, FD);
    cast_pad_bf16<<<cdiv((long long)NC * FCP, 256), T, 0, stream>>>(cell_x, cx_bf, NC, FC, NC, FCP);
    cast_pad_bf16<<<cdiv((long long)FD * 256, 256), T, 0, stream>>>(dW1, dW1b, FD, 256, FD, 256);
    cast_pad_bf16<<<cdiv((long long)FCP * 1024, 256), T, 0, stream>>>(cW1, cW1b, FC, 1024, FCP, 1024);
    cast_pad_bf16<<<cdiv((long long)1024 * 256, 256), T, 0, stream>>>(cW2, cW2b, 1024, 256, 1024, 256);
    cast_pad_bf16<<<cdiv((long long)256 * 256, 256), T, 0, stream>>>(gdW, gdWb, 256, 256, 256, 256);
    cast_pad_bf16<<<cdiv((long long)256 * 1024, 256), T, 0, stream>>>(g1W, g1Wb, 256, 1024, 256, 1024);
    cast_pad_bf16<<<cdiv((long long)1024 * 256, 256), T, 0, stream>>>(g2W, g2Wb, 1024, 256, 1024, 256);
    cast_pad_bf16<<<cdiv((long long)512 * 512, 256), T, 0, stream>>>(rW1, rW1b, 512, 512, 512, 512);
    cast_pad_bf16<<<cdiv((long long)512 * 512, 256), T, 0, stream>>>(rW2, rW2b, 512, 512, 512, 512);

    // ---- feature transforms ----
    gemm(ACT_RELU, dx_bf, FD,   dW1b, db1, nullptr, d_bf,  ND, 256,  FD);    // drug [ND,256]
    gemm(ACT_RELU, cx_bf, FCP,  cW1b, cb1, nullptr, c1_bf, NC, 1024, FCP);   // cell [NC,1024]
    gemm(ACT_RELU, c1_bf, 1024, cW2b, cb2, nullptr, c_bf,  NC, 256,  1024);  // cell [NC,256]

    // ---- drug GAT (256 -> 256) ----
    gemm(ACT_NONE, d_bf, 256, gdWb, nullptr, dxw_f, nullptr, ND, 256, 256);
    gat(d_ei, ED, ND, 256, gdas, gdad, gdb, dxw_f, mx_d, dn_d, as_d, ad_d, eb_d, agg_d, dfin_bf);

    // ---- cell GAT1 (256 -> 1024) ----
    gemm(ACT_NONE, c_bf, 256, g1Wb, nullptr, xw1_f, nullptr, NC, 1024, 256);
    gat(c_ei, EC, NC, 1024, g1as, g1ad, g1b, xw1_f, mx_c, dn_c, as_c, ad_c, eb_c, agg1, c1g_bf);

    // ---- cell GAT2 (1024 -> 256) ----
    gemm(ACT_NONE, c1g_bf, 1024, g2Wb, nullptr, xw2_f, nullptr, NC, 256, 1024);
    gat(c_ei, EC, NC, 256, g2as, g2ad, g2b, xw2_f, mx_c, dn_c, as_c, ad_c, eb_c, agg2, cfin_bf);

    // ---- pair gather + regression head ----
    gather_concat<<<cdiv((long long)NB * 512, 256), T, 0, stream>>>(dfin_bf, cfin_bf, d_idx, c_idx, comb, NB);
    gemm(ACT_ELU, comb,  512, rW1b, rb1, nullptr, h1_bf, NB, 512, 512);
    gemm(ACT_ELU, h1_bf, 512, rW2b, rb2, h2_f,  nullptr, NB, 512, 512);
    head_out<<<cdiv(NB, 8), T, 0, stream>>>(h2_f, rW3, rb3, out, NB, 512);
}
